// JinaPairTraining_472446403305
// MI455X (gfx1250) — matrix operand order repaired and checked by reading, not compiled, not run on hardware
//
#include <hip/hip_runtime.h>
#include <hip/hip_bf16.h>
#include <math.h>

typedef __bf16 bf16;
typedef __attribute__((ext_vector_type(16))) __bf16 v16bf;
typedef __attribute__((ext_vector_type(8)))  __bf16 v8bf;
typedef __attribute__((ext_vector_type(4)))  __bf16 v4bf;
typedef __attribute__((ext_vector_type(8)))  float  v8f;
typedef __attribute__((ext_vector_type(4)))  float  v4f;

#define B_  32
#define T_  256
#define D_  128
#define TAU 0.02f
#define NEG_BIG (-3.4e38f)

// ---- WMMA fragment loaders (contiguous k = 16*(lane/16) + e in both A and B, wave32) ----------

// A-matrix 16x32 bf16: lane L holds row M = L%16; kbase = (L/16)*16;
// elements 0..15 -> K = kbase+e (matches B fragment order)
__device__ __forceinline__ v16bf load_a_frag(const bf16* base, int row, int kc, int lane) {
    const int kbase = (lane >> 4) * 16;
    const bf16* p = base + row * D_ + kc * 32;
    v8bf lo = *(const v8bf*)(p + kbase);
    v8bf hi = *(const v8bf*)(p + 8 + kbase);
    v16bf a;
#pragma unroll
    for (int e = 0; e < 8; ++e) { a[e] = lo[e]; a[e + 8] = hi[e]; }
    return a;
}

// B-matrix 32x16 bf16: lane L holds column N = L%16; K = (L/16)*16 + e
// (B[k][n] = P[n][k] -> 16 contiguous dims of token row "row")
__device__ __forceinline__ v16bf load_b_frag(const bf16* base, int row, int kc, int lane) {
    const bf16* p = base + row * D_ + kc * 32 + (lane >> 4) * 16;
    return *(const v16bf*)p;   // 32B, 32B-aligned
}

// fp32 -> bf16 hi/lo split of a float4
__device__ __forceinline__ void split4(v4f x, v4bf* h, v4bf* l) {
#pragma unroll
    for (int e = 0; e < 4; ++e) {
        bf16 hh = (bf16)x[e];
        (*h)[e] = hh;
        (*l)[e] = (bf16)(x[e] - (float)hh);
    }
}

// ---- Kernel A: late-interaction numerator S_num[b][c] ----------------------
// One workgroup per (b,c); 256 threads = 8 waves; 256KB dynamic LDS staging.
// LDS (256KB) limits us to 1 WG per WGP = 2 waves/SIMD; tell the compiler so
// it can use ~half the VGPR file instead of spilling the A fragments.
__global__ void __launch_bounds__(256)
__attribute__((amdgpu_waves_per_eu(2)))
late_sim_kernel(const float* __restrict__ qmulti,
                const float* __restrict__ pmulti,
                const unsigned char* __restrict__ qmask,
                const unsigned char* __restrict__ pmask,
                float* __restrict__ s_num) {
    extern __shared__ char smem[];
    bf16* qhi = (bf16*)smem;                 // [T_][D_]
    bf16* qlo = qhi + T_ * D_;
    bf16* phi = qlo + T_ * D_;
    bf16* plo = phi + T_ * D_;
    unsigned char* qm = (unsigned char*)(plo + T_ * D_);   // [T_]
    unsigned char* pm = qm + T_;                            // [T_]
    float* acc = (float*)(pm + T_);

    const int t = threadIdx.x;          // 0..255
    const int b = blockIdx.y;
    const int c = blockIdx.x;

    // ---- stage fp32 -> bf16 hi/lo split into LDS (float4 vectorized) ----
    const v4f* qsrc4 = (const v4f*)(qmulti + (size_t)b * T_ * D_);
    const v4f* psrc4 = (const v4f*)(pmulti + (size_t)c * T_ * D_);
    for (int i = t; i < (T_ * D_) / 4; i += 256) {
        v4bf h, l;
        split4(qsrc4[i], &h, &l);
        *(v4bf*)(qhi + 4 * i) = h;
        *(v4bf*)(qlo + 4 * i) = l;
        split4(psrc4[i], &h, &l);
        *(v4bf*)(phi + 4 * i) = h;
        *(v4bf*)(plo + 4 * i) = l;
    }
    if (t < T_) { qm[t] = qmask[b * T_ + t]; pm[t] = pmask[c * T_ + t]; }
    if (t == 0) *acc = 0.0f;
    __syncthreads();

    const int lane = t & 31;
    const int wave = t >> 5;            // 0..7
    const int col  = lane & 15;
    const int qt0  = wave * 2;
    const int qt1  = wave * 2 + 1;

    // p-mask bits for this lane's column, all 16 p-tiles, hoisted
    unsigned pbits = 0;
#pragma unroll
    for (int pt = 0; pt < 16; ++pt)
        pbits |= (pm[pt * 16 + col] ? 1u : 0u) << pt;

    // hoist A fragments: both tiles' hi + tile0's lo (96 VGPRs live);
    // tile1's lo is cheap to re-fetch from LDS per k-chunk.
    v16bf a0h[4], a0l[4], a1h[4];
#pragma unroll
    for (int kc = 0; kc < 4; ++kc) {
        a0h[kc] = load_a_frag(qhi, qt0 * 16 + col, kc, lane);
        a0l[kc] = load_a_frag(qlo, qt0 * 16 + col, kc, lane);
        a1h[kc] = load_a_frag(qhi, qt1 * 16 + col, kc, lane);
    }

    float m0[8], m1[8];                 // per-lane running max (lane = column)
#pragma unroll
    for (int r = 0; r < 8; ++r) { m0[r] = NEG_BIG; m1[r] = NEG_BIG; }

    for (int pt = 0; pt < 16; ++pt) {
        const int prow = pt * 16 + col;
        v8f c0 = {}, c1 = {};
#pragma unroll
        for (int kc = 0; kc < 4; ++kc) {
            v16bf bh = load_b_frag(phi, prow, kc, lane);
            v16bf bl = load_b_frag(plo, prow, kc, lane);
            v16bf a1l = load_a_frag(qlo, qt1 * 16 + col, kc, lane);
            // bf16x3: hi*hi + hi*lo + lo*hi, two independent chains
            c0 = __builtin_amdgcn_wmma_f32_16x16x32_bf16(false, a0h[kc], false, bh, (short)0, c0, false, false);
            c1 = __builtin_amdgcn_wmma_f32_16x16x32_bf16(false, a1h[kc], false, bh, (short)0, c1, false, false);
            c0 = __builtin_amdgcn_wmma_f32_16x16x32_bf16(false, a0h[kc], false, bl, (short)0, c0, false, false);
            c1 = __builtin_amdgcn_wmma_f32_16x16x32_bf16(false, a1h[kc], false, bl, (short)0, c1, false, false);
            c0 = __builtin_amdgcn_wmma_f32_16x16x32_bf16(false, a0l[kc], false, bh, (short)0, c0, false, false);
            c1 = __builtin_amdgcn_wmma_f32_16x16x32_bf16(false, a1l,     false, bh, (short)0, c1, false, false);
        }
        // branchless mask fold (p-mask is per column == per lane)
        const bool pv = (pbits >> pt) & 1u;
#pragma unroll
        for (int r = 0; r < 8; ++r) {
            m0[r] = fmaxf(m0[r], pv ? c0[r] : NEG_BIG);
            m1[r] = fmaxf(m1[r], pv ? c1[r] : NEG_BIG);
        }
    }

    // reduce max across the 16-lane column group
#pragma unroll
    for (int off = 8; off >= 1; off >>= 1) {
#pragma unroll
        for (int r = 0; r < 8; ++r) {
            m0[r] = fmaxf(m0[r], __shfl_xor(m0[r], off, 32));
            m1[r] = fmaxf(m1[r], __shfl_xor(m1[r], off, 32));
        }
    }
    // rows: lanes 0-15 hold rows qt*16+0..7 ; lanes 16-31 rows qt*16+8..15
    if (col == 0) {
        const int half = lane >> 4;
        float s = 0.0f;
#pragma unroll
        for (int r = 0; r < 8; ++r) {
            if (qm[qt0 * 16 + half * 8 + r]) s += m0[r];
            if (qm[qt1 * 16 + half * 8 + r]) s += m1[r];
        }
        atomicAdd(acc, s);              // ds_add_f32
    }
    __syncthreads();
    if (t == 0) s_num[b * B_ + c] = *acc;   // numerator; /t_i in loss kernel
}

// ---- Kernel B: dense 32x32 logits via the same bf16x3 WMMA path ------------
__global__ void dense_sim_kernel(const float* __restrict__ qs,
                                 const float* __restrict__ ps,
                                 float* __restrict__ dense) {
    __shared__ bf16 qhi[B_ * D_], qlo[B_ * D_], phi[B_ * D_], plo[B_ * D_];
    const int t = threadIdx.x;          // 0..127, 4 waves
    const v4f* qs4 = (const v4f*)qs;
    const v4f* ps4 = (const v4f*)ps;
    for (int i = t; i < (B_ * D_) / 4; i += 128) {
        v4bf h, l;
        split4(qs4[i], &h, &l);
        *(v4bf*)(qhi + 4 * i) = h;
        *(v4bf*)(qlo + 4 * i) = l;
        split4(ps4[i], &h, &l);
        *(v4bf*)(phi + 4 * i) = h;
        *(v4bf*)(plo + 4 * i) = l;
    }
    __syncthreads();

    const int lane = t & 31;
    const int wave = t >> 5;
    const int tm = wave >> 1, tn = wave & 1;
    const int col = lane & 15;

    v8f cacc = {};
#pragma unroll
    for (int kc = 0; kc < 4; ++kc) {
        v16bf ah = load_a_frag(qhi, tm * 16 + col, kc, lane);
        v16bf al = load_a_frag(qlo, tm * 16 + col, kc, lane);
        v16bf bh = load_b_frag(phi, tn * 16 + col, kc, lane);
        v16bf bl = load_b_frag(plo, tn * 16 + col, kc, lane);
        cacc = __builtin_amdgcn_wmma_f32_16x16x32_bf16(false, ah, false, bh, (short)0, cacc, false, false);
        cacc = __builtin_amdgcn_wmma_f32_16x16x32_bf16(false, ah, false, bl, (short)0, cacc, false, false);
        cacc = __builtin_amdgcn_wmma_f32_16x16x32_bf16(false, al, false, bh, (short)0, cacc, false, false);
    }
    const int row0 = tm * 16 + (lane >> 4) * 8;
#pragma unroll
    for (int r = 0; r < 8; ++r)
        dense[(row0 + r) * B_ + tn * 16 + col] = cacc[r];
}

// ---- Kernel C: CE (dense + late) + KL, one wave32 --------------------------
__global__ void loss_kernel(const float* __restrict__ dense,
                            const float* __restrict__ s_num,
                            const unsigned char* __restrict__ qmask,
                            float* __restrict__ out) {
    const int j = threadIdx.x;          // 0..31, lane j owns column j
    int cnt = 0;
    for (int k = 0; k < T_; ++k) cnt += qmask[j * T_ + k] ? 1 : 0;
    if (cnt < 1) cnt = 1;               // t_i for row j lives on lane j

    const float inv_tau = 1.0f / TAU;
    const float eps = 1e-8f;
    float ce_d = 0.0f, ce_l = 0.0f, kl = 0.0f;

    for (int i = 0; i < B_; ++i) {
        const float ti = (float)__shfl(cnt, i, 32);
        float xd = dense[i * B_ + j] * inv_tau;
        float xl = (s_num[i * B_ + j] / ti) * inv_tau;

        float md = xd, ml = xl;
#pragma unroll
        for (int off = 16; off >= 1; off >>= 1) {
            md = fmaxf(md, __shfl_xor(md, off, 32));
            ml = fmaxf(ml, __shfl_xor(ml, off, 32));
        }
        float ed = expf(xd - md), el = expf(xl - ml);
        float sd = ed, sl = el;
#pragma unroll
        for (int off = 16; off >= 1; off >>= 1) {
            sd += __shfl_xor(sd, off, 32);
            sl += __shfl_xor(sl, off, 32);
        }
        // diagonal CE terms (all lanes identical after broadcast)
        ce_d += md + logf(sd) - __shfl(xd, i, 32);
        ce_l += ml + logf(sl) - __shfl(xl, i, 32);
        // KL row term
        float dp = ed / sd, lp = el / sl;
        float term = dp * logf((dp + eps) / (lp + eps));
#pragma unroll
        for (int off = 16; off >= 1; off >>= 1) term += __shfl_xor(term, off, 32);
        kl += term;
    }
    const float single = ce_d / (float)B_;
    const float multi  = ce_l / (float)B_;
    const float klm    = kl   / (float)B_;
    if (j == 0) {
        out[0] = single + multi + klm;
        out[1] = single;
        out[2] = multi;
        out[3] = klm;
    }
}

// ---- Host launcher ----------------------------------------------------------
extern "C" void kernel_launch(void* const* d_in, const int* in_sizes, int n_in,
                              void* d_out, int out_size, void* d_ws, size_t ws_size,
                              hipStream_t stream) {
    (void)in_sizes; (void)n_in; (void)out_size; (void)ws_size;
    const float* query_single = (const float*)d_in[0];
    const float* pos_single   = (const float*)d_in[1];
    const float* query_multi  = (const float*)d_in[2];
    const float* pos_multi    = (const float*)d_in[3];
    const unsigned char* q_mask = (const unsigned char*)d_in[4];
    const unsigned char* p_mask = (const unsigned char*)d_in[5];

    float* s_num = (float*)d_ws;                 // [32*32]
    float* dense = s_num + B_ * B_;              // [32*32]
    float* out   = (float*)d_out;                // [4]

    // 4 x (256*128) bf16 staging + 2 masks + 1 accumulator
    const size_t smem_bytes = 4u * T_ * D_ * sizeof(bf16) + 2u * T_ + 16u;
    static_assert(4u * T_ * D_ * sizeof(bf16) + 2u * T_ + 16u <= 320u * 1024u, "LDS budget");
    hipFuncSetAttribute(reinterpret_cast<const void*>(late_sim_kernel),
                        hipFuncAttributeMaxDynamicSharedMemorySize, (int)smem_bytes);

    dim3 grid(B_, B_);
    late_sim_kernel<<<grid, 256, smem_bytes, stream>>>(query_multi, pos_multi,
                                                       q_mask, p_mask, s_num);
    dense_sim_kernel<<<1, 128, 0, stream>>>(query_single, pos_single, dense);
    loss_kernel<<<1, 32, 0, stream>>>(dense, s_num, q_mask, out);
}
